// LocalWindowAttention_83940840833457
// MI455X (gfx1250) — matrix operand order repaired and checked by reading, not compiled, not run on hardware
//
#include <hip/hip_runtime.h>
#include <hip/hip_bf16.h>

typedef __attribute__((ext_vector_type(16))) _Float16     v16h;
typedef __attribute__((ext_vector_type(8)))  _Float16     v8h;
typedef __attribute__((ext_vector_type(8)))  float        v8f;
typedef __attribute__((ext_vector_type(4)))  unsigned int u32x4;
typedef __attribute__((ext_vector_type(8)))  int          i32x8;
typedef __attribute__((ext_vector_type(4)))  int          i32x4;

__device__ __forceinline__ v16h cat8(v8h lo, v8h hi) {
    return __builtin_shufflevector(lo, hi, 0,1,2,3,4,5,6,7,8,9,10,11,12,13,14,15);
}

__device__ __forceinline__ v8f wmma16(v16h a, v16h b, v8f c) {
    return __builtin_amdgcn_wmma_f32_16x16x32_f16(false, a, false, b, (short)0, c, false, false);
}

// ---------------------------------------------------------------------------
// Tensor Data Mover: DMA a 2D tile (tile_k x tile_rows of f16, row stride
// stride_elems) from global memory into LDS (packed rows of tile_k halfs).
// D# layout per CDNA5 ISA (S8.3/8.4): count=1, type=2,
// data_size=1 (2 bytes). Issued by one wave; tracked by TENSORcnt (in-order).
// ---------------------------------------------------------------------------
__device__ __forceinline__ void tdm_load_tile_f16(unsigned lds_off,
                                                  const _Float16* gptr,
                                                  unsigned tile_k, unsigned tile_rows,
                                                  unsigned tensor_d0, unsigned tensor_d1,
                                                  unsigned long long stride_elems)
{
    unsigned long long ga = (unsigned long long)(size_t)gptr;
    u32x4 g0;
    g0[0] = 1u;                                                // count=1, user mode
    g0[1] = lds_off;                                           // lds_addr [63:32]
    g0[2] = (unsigned)(ga & 0xffffffffu);                      // global_addr lo32
    g0[3] = (unsigned)((ga >> 32) & 0x01ffffffu) | (2u << 30); // addr[56:32] | type=2
    i32x8 g1;
    g1[0] = (int)(1u << 16);                                   // data_size=1 (2B), mask=0
    g1[1] = (int)((tensor_d0 & 0xffffu) << 16);                // tensor_dim0 lo16
    g1[2] = (int)(((tensor_d0 >> 16) & 0xffffu) | ((tensor_d1 & 0xffffu) << 16));
    g1[3] = (int)(((tensor_d1 >> 16) & 0xffffu) | ((tile_k & 0xffffu) << 16)); // tile_dim0
    g1[4] = (int)(tile_rows & 0xffffu);                        // tile_dim1 (tile_dim2=0)
    g1[5] = (int)(stride_elems & 0xffffffffu);                 // tensor_dim0_stride lo32
    g1[6] = (int)((stride_elems >> 32) & 0xffffu);             // stride hi16
    g1[7] = 0;
    i32x4 z4 = {0, 0, 0, 0};
#if __clang_major__ >= 23
    i32x8 z8 = {0, 0, 0, 0, 0, 0, 0, 0};
    __builtin_amdgcn_tensor_load_to_lds(g0, g1, z4, z4, z8, 0);
#else
    __builtin_amdgcn_tensor_load_to_lds(g0, g1, z4, z4, 0);
#endif
}

// ---------------------------------------------------------------------------
// Prep kernels: f32 -> f16 convert, and convert + transpose for weights
// ---------------------------------------------------------------------------
__global__ __launch_bounds__(256)
void cvt_f16_kernel(const float* __restrict__ in, _Float16* __restrict__ out)
{
    size_t i = ((size_t)blockIdx.x * 256 + threadIdx.x) * 8;
    float4 a = *(const float4*)&in[i];
    float4 b = *(const float4*)&in[i + 4];
    _Float16 t[8] = {(_Float16)a.x, (_Float16)a.y, (_Float16)a.z, (_Float16)a.w,
                     (_Float16)b.x, (_Float16)b.y, (_Float16)b.z, (_Float16)b.w};
    *(uint4*)&out[i] = *(const uint4*)t;
}

// W[k][n] (f32, KxN) -> Wt[n][k] (f16, NxK)
__global__ __launch_bounds__(256)
void transpose_cvt_kernel(const float* __restrict__ W, _Float16* __restrict__ Wt,
                          int K, int N)
{
    __shared__ _Float16 tile[32][33];
    const int k0 = blockIdx.y * 32, n0 = blockIdx.x * 32;
    const int tx = threadIdx.x & 31, ty = threadIdx.x >> 5;   // 32 x 8
    #pragma unroll
    for (int i = 0; i < 32; i += 8)
        tile[ty + i][tx] = (_Float16)W[(size_t)(k0 + ty + i) * N + n0 + tx];
    __syncthreads();
    #pragma unroll
    for (int i = 0; i < 32; i += 8)
        Wt[(size_t)(n0 + ty + i) * K + k0 + tx] = tile[tx][ty + i];
}

// ---------------------------------------------------------------------------
// GEMM: C[M,N] = A[M,K](f16) * Bt[N,K](f16, pre-transposed) + bias[N]
// Block 128 thr = 4 waves; tile 64x64, K-step 64.
// Double-buffered LDS; tiles staged by the TDM, overlapped with WMMA:
//   iter i: wave0 issues DMA(i+1) -> buf[(i+1)&1], waits TENSORcnt<=2
//   (in-order completion => tile i resident), all waves compute buf[i&1]
//   while tile i+1 streams in.
// ---------------------------------------------------------------------------
template <typename TC>
__global__ __launch_bounds__(128)
void gemm_wmma_tdm_kernel(const _Float16* __restrict__ A,
                          const _Float16* __restrict__ Bt,
                          const float* __restrict__ bias,
                          TC* __restrict__ C, int M, int N, int K)
{
    __shared__ __align__(16) _Float16 lds_a[2][64][64];   // [buf][m][k]
    __shared__ __align__(16) _Float16 lds_b[2][64][64];   // [buf][n][k]

    const int tid  = threadIdx.x;
    const int wave = tid >> 5;
    const int lane = tid & 31;
    const int l16  = lane & 15;
    const bool hiH = lane >= 16;

    const int m_block = blockIdx.y * 64;
    const int n_block = blockIdx.x * 64;
    const int m0      = wave * 16;
    const int niter   = K >> 6;

    const _Float16* Ab = A  + (size_t)m_block * K;
    const _Float16* Bb = Bt + (size_t)n_block * K;

    v8f acc[4] = {};

    // prologue: kick off tile 0
    if (tid < 32) {
        tdm_load_tile_f16((unsigned)(size_t)&lds_a[0][0][0], Ab,
                          64, 64, (unsigned)K, (unsigned)M, (unsigned)K);
        tdm_load_tile_f16((unsigned)(size_t)&lds_b[0][0][0], Bb,
                          64, 64, (unsigned)K, (unsigned)N, (unsigned)K);
    }

    for (int it = 0; it < niter; ++it) {
        const int cur = it & 1;
        if (tid < 32) {
            if (it + 1 < niter) {
                const int nxt = (it + 1) & 1;
                const int k1  = (it + 1) << 6;
                tdm_load_tile_f16((unsigned)(size_t)&lds_a[nxt][0][0], Ab + k1,
                                  64, 64, (unsigned)K, (unsigned)M, (unsigned)K);
                tdm_load_tile_f16((unsigned)(size_t)&lds_b[nxt][0][0], Bb + k1,
                                  64, 64, (unsigned)K, (unsigned)N, (unsigned)K);
                __builtin_amdgcn_s_wait_tensorcnt(2);  // tile `it` resident
            } else {
                __builtin_amdgcn_s_wait_tensorcnt(0);  // last tile resident
            }
        }
        __syncthreads();                               // buf[cur] visible to all

        const int am = m0 + l16;
        #pragma unroll
        for (int ks = 0; ks < 2; ++ks) {
            const int kc  = ks * 32;
            const int ac0 = kc + (hiH ? 8 : 0);
            v16h afrag = cat8(*(const v8h*)&lds_a[cur][am][ac0],
                              *(const v8h*)&lds_a[cur][am][ac0 + 16]);
            const int bk0 = kc + (hiH ? 8 : 0);
            #pragma unroll
            for (int nt = 0; nt < 4; ++nt) {
                const int bn = nt * 16 + l16;
                v16h bfrag = cat8(*(const v8h*)&lds_b[cur][bn][bk0],
                                  *(const v8h*)&lds_b[cur][bn][bk0 + 16]);
                acc[nt] = wmma16(afrag, bfrag, acc[nt]);
            }
        }
        __syncthreads();           // buf[cur] free for DMA reuse next iteration
    }

    const int mrow = m_block + m0 + (hiH ? 8 : 0);
    #pragma unroll
    for (int nt = 0; nt < 4; ++nt) {
        const int n = n_block + nt * 16 + l16;
        const float bv = bias[n];
        #pragma unroll
        for (int i = 0; i < 8; ++i)
            C[(size_t)(mrow + i) * N + n] = (TC)(acc[nt][i] + bv);
    }
}

// ---------------------------------------------------------------------------
// Windowed attention (f16 qkv in, f16 out). Block per (head, window, batch).
// qkv row = token, 3072 cols = [s(3)][h(16)][d(64)].
// ---------------------------------------------------------------------------
__global__ __launch_bounds__(128)
void attn_wmma_kernel(const _Float16* __restrict__ qkv,
                      const float* __restrict__ rel_bias,
                      _Float16* __restrict__ outbuf)
{
    __shared__ __align__(16) _Float16 lds_q [64][64];   // [t][d]
    __shared__ __align__(16) _Float16 lds_k [64][64];   // [t][d]
    __shared__ __align__(16) _Float16 lds_vt[64][64];   // [d][t]
    __shared__ __align__(16) _Float16 lds_p [64][64];   // [q][t]
    __shared__ float lds_bias[128];

    const int h = blockIdx.x, w = blockIdx.y, b = blockIdx.z;
    const int tid  = threadIdx.x;
    const int wave = tid >> 5;
    const int lane = tid & 31;
    const int l16  = lane & 15;
    const bool hiH = lane >= 16;

    const size_t rowbase = (size_t)(b * 64 + w) * 64;
    const _Float16* qp = qkv + rowbase * 3072 + (size_t)h * 64;

    // stage q, k (straight b128 copies) and v transposed
    #pragma unroll
    for (int it = 0; it < 4; ++it) {
        int c  = tid + it * 128;          // 512 chunks of 8 halfs per tile
        int r  = c >> 3;
        int d0 = (c & 7) * 8;
        const _Float16* base = qp + (size_t)r * 3072 + d0;
        *(uint4*)&lds_q[r][d0] = *(const uint4*)(base);
        *(uint4*)&lds_k[r][d0] = *(const uint4*)(base + 1024);
        uint4 vv = *(const uint4*)(base + 2048);
        _Float16 tmp[8]; *(uint4*)tmp = vv;
        #pragma unroll
        for (int j = 0; j < 8; ++j) lds_vt[d0 + j][r] = tmp[j];
    }
    if (tid < 127) lds_bias[tid] = rel_bias[tid * 16 + h];
    __syncthreads();

    // S = q @ k^T
    const int m0 = wave * 16;
    v8f acc[4] = {};
    #pragma unroll
    for (int ks = 0; ks < 2; ++ks) {
        int kc  = ks * 32;
        int ac0 = kc + (hiH ? 8 : 0);
        v16h afrag = cat8(*(const v8h*)&lds_q[m0 + l16][ac0],
                          *(const v8h*)&lds_q[m0 + l16][ac0 + 16]);
        int bk0 = kc + (hiH ? 8 : 0);
        #pragma unroll
        for (int nt = 0; nt < 4; ++nt) {
            int bn = nt * 16 + l16;
            v16h bfrag = cat8(*(const v8h*)&lds_k[bn][bk0],
                              *(const v8h*)&lds_k[bn][bk0 + 16]);
            acc[nt] = wmma16(afrag, bfrag, acc[nt]);
        }
    }

    // scale + relative bias + softmax (row = 16 lanes x 4 tiles)
    const float scale = 0.125f;
    const int rowoff = hiH ? 8 : 0;
    #pragma unroll
    for (int i = 0; i < 8; ++i) {
        int qi = m0 + i + rowoff;
        float vals[4];
        float mx = -3.0e38f;
        #pragma unroll
        for (int nt = 0; nt < 4; ++nt) {
            int ki = nt * 16 + l16;
            float v = acc[nt][i] * scale + lds_bias[qi - ki + 63];
            vals[nt] = v; mx = fmaxf(mx, v);
        }
        #pragma unroll
        for (int off = 8; off >= 1; off >>= 1)
            mx = fmaxf(mx, __shfl_xor(mx, off, 32));
        float s = 0.f;
        #pragma unroll
        for (int nt = 0; nt < 4; ++nt) { float e = __expf(vals[nt] - mx); vals[nt] = e; s += e; }
        #pragma unroll
        for (int off = 8; off >= 1; off >>= 1)
            s += __shfl_xor(s, off, 32);
        float inv = 1.0f / s;
        #pragma unroll
        for (int nt = 0; nt < 4; ++nt)
            lds_p[qi][nt * 16 + l16] = (_Float16)(vals[nt] * inv);
    }
    // P rows are wave-private; per-wave in-order DScnt suffices.

    // O = P @ V
    v8f oacc[4] = {};
    #pragma unroll
    for (int ks = 0; ks < 2; ++ks) {
        int kc  = ks * 32;
        int ac0 = kc + (hiH ? 8 : 0);
        v16h afrag = cat8(*(const v8h*)&lds_p[m0 + l16][ac0],
                          *(const v8h*)&lds_p[m0 + l16][ac0 + 16]);
        int bk0 = kc + (hiH ? 8 : 0);
        #pragma unroll
        for (int nt = 0; nt < 4; ++nt) {
            int bn = nt * 16 + l16;
            v16h bfrag = cat8(*(const v8h*)&lds_vt[bn][bk0],
                              *(const v8h*)&lds_vt[bn][bk0 + 16]);
            oacc[nt] = wmma16(afrag, bfrag, oacc[nt]);
        }
    }

    const size_t orow = rowbase + m0 + rowoff;
    #pragma unroll
    for (int nt = 0; nt < 4; ++nt) {
        int d = nt * 16 + l16;
        #pragma unroll
        for (int i = 0; i < 8; ++i)
            outbuf[(orow + i) * 1024 + h * 64 + d] = (_Float16)oacc[nt][i];
    }
}

// ---------------------------------------------------------------------------
extern "C" void kernel_launch(void* const* d_in, const int* in_sizes, int n_in,
                              void* d_out, int out_size, void* d_ws, size_t ws_size,
                              hipStream_t stream)
{
    const float* x        = (const float*)d_in[0];   // (8,4096,1024)
    const float* qkv_w    = (const float*)d_in[1];   // (1024,3072)
    const float* qkv_b    = (const float*)d_in[2];   // (3072,)
    const float* proj_w   = (const float*)d_in[3];   // (1024,1024)
    const float* proj_b   = (const float*)d_in[4];   // (1024,)
    const float* rel_bias = (const float*)d_in[5];   // (127,16)

    const int M = 8 * 4096;      // 32768 tokens
    const int D = 1024;

    _Float16* xh     = (_Float16*)d_ws;                 // M x D
    _Float16* wqkvt  = xh     + (size_t)M * D;          // 3D x D (transposed)
    _Float16* wprojt = wqkvt  + (size_t)3 * D * D;      // D x D  (transposed)
    _Float16* qkvh   = wprojt + (size_t)D * D;          // M x 3D
    _Float16* attnh  = qkvh   + (size_t)M * 3 * D;      // M x D
    float*    out    = (float*)d_out;

    // 0) precision prep: x -> f16; weights -> f16 transposed [n][k]
    cvt_f16_kernel<<<(size_t)M * D / (256 * 8), 256, 0, stream>>>(x, xh);
    transpose_cvt_kernel<<<dim3(3 * D / 32, D / 32), 256, 0, stream>>>(qkv_w, wqkvt, D, 3 * D);
    transpose_cvt_kernel<<<dim3(D / 32, D / 32), 256, 0, stream>>>(proj_w, wprojt, D, D);

    // 1) QKV projection (f16 out)
    gemm_wmma_tdm_kernel<_Float16><<<dim3(3 * D / 64, M / 64), 128, 0, stream>>>(
        xh, wqkvt, qkv_b, qkvh, M, 3 * D, D);

    // 2) windowed attention: one block per (head, window, batch)
    attn_wmma_kernel<<<dim3(16, 64, 8), 128, 0, stream>>>(qkvh, rel_bias, attnh);

    // 3) output projection (f32 out)
    gemm_wmma_tdm_kernel<float><<<dim3(D / 64, M / 64), 128, 0, stream>>>(
        attnh, wprojt, proj_b, out, M, D, D);
}
